// GraphAttentionLayer_88880053224022
// MI455X (gfx1250) — matrix operand order repaired and checked by reading, not compiled, not run on hardware
//
#include <hip/hip_runtime.h>
#include <hip/hip_bf16.h>
#include <stdint.h>

// ---------------------------------------------------------------------------
// GAT layer for MI455X (gfx1250, wave32, WMMA).
//  K1: Wh = h@W (bf16 WMMA), emits Wh^T bf16 (K-contiguous) + s1/s2 scores.
//  K2: fused masked-softmax + attention@Wh, flash-style. Pass 1 is max-only
//      (no exp); the softmax denominator is accumulated for free in pass 2
//      from the unnormalized P' tiles, and applied in the epilogue.
// Roofline: adj stream 134MB @23.3TB/s ~ 6us floor; bf16 WMMA ~3us and the
// minimal 34M exps hide underneath. Both softmax passes over a WG's 128KB adj
// slice hit the 192MB L2.
// ---------------------------------------------------------------------------

typedef __attribute__((ext_vector_type(16))) __bf16 v16bf;
typedef __attribute__((ext_vector_type(8)))  float  v8f;

#define GAT_ALPHA 0.2f
#define GAT_NEG_BIG -9.0e15f

union V16U { v16bf v; uint4 q[2]; };

// ===========================================================================
// Kernel 1: Wh = h @ W   (16 rows per block, full Fout=256 width)
// block = 256 threads = 8 waves; wave w owns Fout columns [32w, 32w+32)
// ===========================================================================
__global__ __launch_bounds__(256)
void gat_wh_kernel(const float* __restrict__ h, const float* __restrict__ W,
                   const float* __restrict__ a,
                   __bf16* __restrict__ whT, float* __restrict__ s1g,
                   float* __restrict__ s2g)
{
    __shared__ __bf16 htile[16 * 264];   // 16 rows x 256 K, pad->264
    __shared__ __bf16 wtile[256 * 40];   // 256 o  x 32 K,  pad->40 (transposed)
    __shared__ float  whtile[16 * 260];  // epilogue staging

    const int tid  = threadIdx.x;
    const int lane = tid & 31;
    const int wv   = tid >> 5;
    const int g    = lane >> 4;      // half-wave group (0/1)
    const int l16  = lane & 15;
    const int b    = blockIdx.x >> 7;
    const int n0   = (blockIdx.x & 127) << 4;

    // stage h tile (16 x 256) as bf16, coalesced
    {
        const float* hp = h + ((size_t)(b * 2048 + n0)) * 256;
        #pragma unroll
        for (int it = 0; it < 16; ++it)
            htile[it * 264 + tid] = (__bf16)hp[it * 256 + tid];
    }

    v8f acc0 = {}; v8f acc1 = {};

    for (int k0 = 0; k0 < 256; k0 += 32) {
        __syncthreads();
        // stage W^T tile: wtile[o*40 + kk] = W[k0+kk][o], coalesced reads
        #pragma unroll
        for (int it = 0; it < 32; ++it)
            wtile[tid * 40 + it] = (__bf16)W[(k0 + it) * 256 + tid];
        __syncthreads();

        // A operand (16x32 bf16): lanes0-15 K0-15, lanes16-31 K16-31
        V16U A;
        A.q[0] = *(const uint4*)&htile[l16 * 264 + k0 + g * 16];
        A.q[1] = *(const uint4*)&htile[l16 * 264 + k0 + g * 16 + 8];

        // B operands (32x16 bf16): lane=col, lanes16-31 hold K16-31
        const int o0 = wv * 32 + l16;
        V16U B0, B1;
        {
            const uint4* b0 = (const uint4*)&wtile[o0 * 40 + g * 16];
            const uint4* b1 = (const uint4*)&wtile[(o0 + 16) * 40 + g * 16];
            B0.q[0] = b0[0]; B0.q[1] = b0[1];
            B1.q[0] = b1[0]; B1.q[1] = b1[1];
        }
        acc0 = __builtin_amdgcn_wmma_f32_16x16x32_bf16(false, A.v, false, B0.v,
                                                       (short)0, acc0, false, false);
        acc1 = __builtin_amdgcn_wmma_f32_16x16x32_bf16(false, A.v, false, B1.v,
                                                       (short)0, acc1, false, false);
    }

    __syncthreads();
    // scatter C (vgpr r -> rows r / r+8) into LDS f32 staging tile
    #pragma unroll
    for (int r = 0; r < 8; ++r) {
        const int m = r + g * 8;
        whtile[m * 260 + wv * 32 + l16]      = acc0[r];
        whtile[m * 260 + wv * 32 + 16 + l16] = acc1[r];
    }
    __syncthreads();

    // Wh^T bf16 store: thread tid == o; n-dimension contiguous (K-contiguous
    // for kernel 2's B operand loads)
    {
        __bf16* wp = whT + ((size_t)(b * 256 + tid)) * 2048 + n0;
        #pragma unroll
        for (int m = 0; m < 16; ++m)
            wp[m] = (__bf16)whtile[m * 260 + tid];
    }

    // s1 = Wh . a[:256], s2 = Wh . a[256:]  (deterministic, f32)
    if (tid < 16) {
        float su1 = 0.f, su2 = 0.f;
        for (int o = 0; o < 256; ++o) {
            const float v = whtile[tid * 260 + o];
            su1 += v * a[o];
            su2 += v * a[256 + o];
        }
        s1g[b * 2048 + n0 + tid] = su1;
        s2g[b * 2048 + n0 + tid] = su2;
    }
}

// ===========================================================================
// Kernel 2: fused masked softmax + attention @ Wh
// block = (b, 16-row i-tile); pass1 = row max only (no exp);
// pass2 = 64-wide j-tiles: unnormalized P' bf16 in LDS (double-buffered) ->
// WMMA; denominator l accumulated per-thread, applied in epilogue.
// ===========================================================================
__global__ __launch_bounds__(256)
void gat_attn_kernel(const int* __restrict__ adj, const __bf16* __restrict__ whT,
                     const float* __restrict__ s1g, const float* __restrict__ s2g,
                     float* __restrict__ out)
{
    __shared__ __bf16 Pb[2][16 * 72];     // P tiles, pad stride 72
    __shared__ float  ls1[16], lm[16], ll[16], linv[16];
    __shared__ float  redm[16][8];

    const int tid  = threadIdx.x;
    const int lane = tid & 31;
    const int wv   = tid >> 5;
    const int g    = lane >> 4;
    const int l16  = lane & 15;
    const int b    = blockIdx.x >> 7;
    const int i0   = (blockIdx.x & 127) << 4;

    if (tid < 16) ls1[tid] = s1g[b * 2048 + i0 + tid];
    __syncthreads();

    const size_t adjBase = ((size_t)b * 2048 + i0) * 2048;

    // ---------------- pass 1: row max only (pure v_max, no exp) -----------
    float m[16];
    #pragma unroll
    for (int i = 0; i < 16; ++i) m[i] = -3.0e38f;

    for (int jb = 0; jb < 2; ++jb) {
        const int j = jb * 1024 + tid * 4;
        const float4 s2v = *(const float4*)(s2g + b * 2048 + j);
        #pragma unroll
        for (int i = 0; i < 16; ++i) {
            const int4 av = *(const int4*)(adj + adjBase + (size_t)i * 2048 + j);
            const float s1v = ls1[i];
            float e0 = s1v + s2v.x; e0 = fmaxf(e0, GAT_ALPHA * e0); e0 = av.x > 0 ? e0 : GAT_NEG_BIG;
            float e1 = s1v + s2v.y; e1 = fmaxf(e1, GAT_ALPHA * e1); e1 = av.y > 0 ? e1 : GAT_NEG_BIG;
            float e2 = s1v + s2v.z; e2 = fmaxf(e2, GAT_ALPHA * e2); e2 = av.z > 0 ? e2 : GAT_NEG_BIG;
            float e3 = s1v + s2v.w; e3 = fmaxf(e3, GAT_ALPHA * e3); e3 = av.w > 0 ? e3 : GAT_NEG_BIG;
            m[i] = fmaxf(m[i], fmaxf(fmaxf(e0, e1), fmaxf(e2, e3)));
        }
    }
    // wave32 butterfly max, then cross-wave combine
    #pragma unroll
    for (int i = 0; i < 16; ++i) {
        #pragma unroll
        for (int off = 16; off >= 1; off >>= 1)
            m[i] = fmaxf(m[i], __shfl_xor(m[i], off, 32));
        if (lane == 0) redm[i][wv] = m[i];
    }
    __syncthreads();
    if (tid < 16) {
        float mm = redm[tid][0];
        #pragma unroll
        for (int w = 1; w < 8; ++w) mm = fmaxf(mm, redm[tid][w]);
        lm[tid] = mm;
    }
    __syncthreads();

    // ---------------- pass 2: P' tiles -> WMMA accumulate -----------------
    v8f acc0 = {}; v8f acc1 = {};
    float lsum = 0.f;                   // this thread's share of sum exp(e-m)
    const int   pi  = tid >> 4;         // P-row this thread fills
    const int   pj  = (tid & 15) * 4;   // 4 P-cols
    const float s1i = ls1[pi];
    const float mi  = lm[pi];
    const int   o0  = wv * 32 + l16;
    const __bf16* wrow0 = whT + ((size_t)(b * 256 + o0)) * 2048;
    const __bf16* wrow1 = whT + ((size_t)(b * 256 + o0 + 16)) * 2048;

    for (int jt = 0; jt < 32; ++jt) {
        const int j0  = jt * 64;
        const int buf = jt & 1;
        {
            const int4   av  = *(const int4*)(adj + adjBase + (size_t)pi * 2048 + j0 + pj);
            const float4 s2v = *(const float4*)(s2g + b * 2048 + j0 + pj);
            __bf16* pp = &Pb[buf][pi * 72 + pj];
            float e, p;
            e = s1i + s2v.x; e = fmaxf(e, GAT_ALPHA * e); p = av.x > 0 ? __expf(e - mi) : 0.f; pp[0] = (__bf16)p; lsum += p;
            e = s1i + s2v.y; e = fmaxf(e, GAT_ALPHA * e); p = av.y > 0 ? __expf(e - mi) : 0.f; pp[1] = (__bf16)p; lsum += p;
            e = s1i + s2v.z; e = fmaxf(e, GAT_ALPHA * e); p = av.z > 0 ? __expf(e - mi) : 0.f; pp[2] = (__bf16)p; lsum += p;
            e = s1i + s2v.w; e = fmaxf(e, GAT_ALPHA * e); p = av.w > 0 ? __expf(e - mi) : 0.f; pp[3] = (__bf16)p; lsum += p;
            if (jt + 1 < 32)  // pull next adj tile toward the caches
                __builtin_prefetch(adj + adjBase + (size_t)pi * 2048 + j0 + 64 + pj, 0, 0);
        }
        __syncthreads();  // double-buffered: one barrier per tile

        #pragma unroll
        for (int kk = 0; kk < 64; kk += 32) {
            V16U A;
            A.q[0] = *(const uint4*)&Pb[buf][l16 * 72 + kk + g * 16];
            A.q[1] = *(const uint4*)&Pb[buf][l16 * 72 + kk + g * 16 + 8];
            const int kb = j0 + kk + g * 16;
            V16U B0, B1;
            {
                const uint4* bp0 = (const uint4*)(wrow0 + kb);
                const uint4* bp1 = (const uint4*)(wrow1 + kb);
                B0.q[0] = bp0[0]; B0.q[1] = bp0[1];
                B1.q[0] = bp1[0]; B1.q[1] = bp1[1];
            }
            acc0 = __builtin_amdgcn_wmma_f32_16x16x32_bf16(false, A.v, false, B0.v,
                                                           (short)0, acc0, false, false);
            acc1 = __builtin_amdgcn_wmma_f32_16x16x32_bf16(false, A.v, false, B1.v,
                                                           (short)0, acc1, false, false);
        }
    }

    // reduce lsum across the 16 lanes sharing row pi (stays in half-wave)
    #pragma unroll
    for (int off = 8; off >= 1; off >>= 1)
        lsum += __shfl_xor(lsum, off, 32);
    if ((tid & 15) == 0) ll[pi] = lsum;
    __syncthreads();
    if (tid < 16) linv[tid] = 1.0f / ll[tid];
    __syncthreads();

    // C layout: vgpr r -> rows r (lanes 0-15) / r+8 (lanes 16-31), col = lane&15
    float* op = out + ((size_t)(b * 2048) + i0) * 256;
    #pragma unroll
    for (int r = 0; r < 8; ++r) {
        const int mrow = r + g * 8;
        const float sc = linv[mrow];
        op[(size_t)mrow * 256 + o0]      = acc0[r] * sc;
        op[(size_t)mrow * 256 + o0 + 16] = acc1[r] * sc;
    }
}

// ===========================================================================
extern "C" void kernel_launch(void* const* d_in, const int* in_sizes, int n_in,
                              void* d_out, int out_size, void* d_ws, size_t ws_size,
                              hipStream_t stream)
{
    const float* h   = (const float*)d_in[0];   // (8,2048,256) f32
    const int*   adj = (const int*)d_in[1];     // (8,2048,2048) i32
    const float* W   = (const float*)d_in[2];   // (256,256) f32
    const float* a   = (const float*)d_in[3];   // (512,1) f32
    float*       out = (float*)d_out;           // (8,2048,256) f32

    char* ws = (char*)d_ws;
    __bf16* whT = (__bf16*)ws;                         // 8 MiB: Wh^T bf16 [b][o][n]
    float*  s1  = (float*)(ws + (size_t)8 * 1024 * 1024);
    float*  s2  = s1 + 8 * 2048;

    gat_wh_kernel<<<1024, 256, 0, stream>>>(h, W, a, whT, s1, s2);
    gat_attn_kernel<<<1024, 256, 0, stream>>>(adj, whT, s1, s2, out);
}
